// GenomeDecoder_59468117180774
// MI455X (gfx1250) — matrix operand order repaired and checked by reading, not compiled, not run on hardware
//
#include <hip/hip_runtime.h>
#include <stdint.h>

#define GL 16
#define NB 16
#define HD 256
#define NCH 1028
#define CHTOT 263168
#define NWG 4
#define GRU_THREADS 128
#define WSLAB_ELEMS 114688            // 28 tiles * 8 kt * 32 lanes * 16 elems (bf16)
#define WSLAB_BYTES 229376
#define ACT_BYTES 8192                // 16x256 bf16
#define SMEM_BYTES (WSLAB_BYTES + 2 * ACT_BYTES)

typedef __attribute__((ext_vector_type(16))) __bf16 v16bf;
typedef __attribute__((ext_vector_type(8)))  float  v8f;

union BFrag { v16bf v; uint4 q[2]; };

#define WMMA_BF16(a, b, c) \
  __builtin_amdgcn_wmma_f32_16x16x32_bf16(false, (a), false, (b), (short)0, (c), false, false)

__device__ __forceinline__ unsigned short f2bf(float f) {
  unsigned u = __builtin_bit_cast(unsigned, f);
  u += 0x7FFFu + ((u >> 16) & 1u);              // round-to-nearest-even
  return (unsigned short)(u >> 16);
}
__device__ __forceinline__ float bf2f(unsigned short h) {
  unsigned u = ((unsigned)h) << 16;
  return __builtin_bit_cast(float, u);
}
__device__ __forceinline__ float sigm(float x) { return 1.0f / (1.0f + __expf(-x)); }

// ---------------------------------------------------------------------------
// Encoder kernels (tiny: ~0.8 GFLOP total) — plain fp32 tiled compute.
// ---------------------------------------------------------------------------

// C[M,N] = act(A[M,K] @ W[N,K]^T + bias[N])
__global__ void k_gemm(const float* __restrict__ A, const float* __restrict__ W,
                       const float* __restrict__ bias, float* __restrict__ C,
                       int M, int N, int K, int relu) {
  __shared__ float As[16][17];
  __shared__ float Ws[16][17];
  int tx = threadIdx.x, ty = threadIdx.y;
  int row = blockIdx.y * 16 + ty;
  int colN = blockIdx.x * 16 + tx;
  float acc = 0.0f;
  for (int k0 = 0; k0 < K; k0 += 16) {
    As[ty][tx] = A[row * K + k0 + tx];
    Ws[ty][tx] = W[(blockIdx.x * 16 + ty) * K + k0 + tx];
    __syncthreads();
#pragma unroll
    for (int kk = 0; kk < 16; ++kk) acc += As[ty][kk] * Ws[tx][kk];
    __syncthreads();
  }
  acc += bias[colN];
  if (relu) acc = fmaxf(acc, 0.0f);
  C[row * N + colN] = acc;
}

// one block per (batch, head); qkv is [256, 768], o is [256, 256]
__global__ void k_attention(const float* __restrict__ qkv, float* __restrict__ o) {
  __shared__ float qs[16][64], ks[16][64], vs[16][64], sc[16][16];
  int b = blockIdx.x >> 2, hd = blockIdx.x & 3;
  int tx = threadIdx.x, ty = threadIdx.y;
  int tid = ty * 16 + tx;
  for (int i = tid; i < 1024; i += 256) {
    int s = i >> 6, d = i & 63;
    int base = (b * 16 + s) * 768 + hd * 64 + d;
    qs[s][d] = qkv[base];
    ks[s][d] = qkv[base + 256];
    vs[s][d] = qkv[base + 512];
  }
  __syncthreads();
  float acc = 0.0f;
#pragma unroll
  for (int d = 0; d < 64; ++d) acc += qs[ty][d] * ks[tx][d];
  sc[ty][tx] = acc * 0.125f;                    // 1/sqrt(64)
  __syncthreads();
  if (tx == 0) {
    float mx = sc[ty][0];
    for (int t2 = 1; t2 < 16; ++t2) mx = fmaxf(mx, sc[ty][t2]);
    float sum = 0.0f;
    for (int t2 = 0; t2 < 16; ++t2) { float e = __expf(sc[ty][t2] - mx); sc[ty][t2] = e; sum += e; }
    float inv = 1.0f / sum;
    for (int t2 = 0; t2 < 16; ++t2) sc[ty][t2] *= inv;
  }
  __syncthreads();
#pragma unroll
  for (int j = 0; j < 4; ++j) {
    int d = tx * 4 + j;
    float a2 = 0.0f;
    for (int t2 = 0; t2 < 16; ++t2) a2 += sc[ty][t2] * vs[t2][d];
    o[(b * 16 + ty) * 256 + hd * 64 + d] = a2;
  }
}

// h[row] = LayerNorm(h[row] + d[row]) * g + b ; one block per row (256 threads)
__global__ void k_add_ln(float* __restrict__ h, const float* __restrict__ d,
                         const float* __restrict__ g, const float* __restrict__ b) {
  __shared__ float s[256];
  int row = blockIdx.x, t = threadIdx.x;
  float x = h[row * 256 + t] + d[row * 256 + t];
  s[t] = x;
  __syncthreads();
  for (int st = 128; st > 0; st >>= 1) { if (t < st) s[t] += s[t + st]; __syncthreads(); }
  float m = s[0] * (1.0f / 256.0f);
  __syncthreads();
  float c = x - m;
  s[t] = c * c;
  __syncthreads();
  for (int st = 128; st > 0; st >>= 1) { if (t < st) s[t] += s[t + st]; __syncthreads(); }
  float var = s[0] * (1.0f / 256.0f);
  h[row * 256 + t] = c * rsqrtf(var + 1e-5f) * g[t] + b[t];
}

__global__ void k_mean(const float* __restrict__ h, float* __restrict__ lat) {
  int b = blockIdx.x, t = threadIdx.x;
  float a = 0.0f;
  for (int s = 0; s < GL; ++s) a += h[(b * GL + s) * HD + t];
  lat[b * HD + t] = a * (1.0f / 16.0f);
}

// ---------------------------------------------------------------------------
// GRU weight packer: fp32 -> bf16, rearranged into per-WG / per-wave WMMA
// B-fragment layout. Slab layout: [wg][q 0..3][t 0..6][kt 0..7][lane 0..31][e 0..15]
//   t: 0..2 = Wih rows (r,z,n), 3..5 = Whh rows (r,z,n), 6 = out_W rows.
//   B-operand of wmma is K x N (32x16): lane&15 -> N, e and lane>>4 -> K (K = kt*32 + 8*(lane>>4) + e, +8 for e >= 8).
// ---------------------------------------------------------------------------
__global__ void k_pack(const float* __restrict__ Wih, const float* __restrict__ Whh,
                       const float* __restrict__ Wout, unsigned short* __restrict__ packW) {
  int idx = blockIdx.x * 256 + threadIdx.x;
  if (idx >= NWG * WSLAB_ELEMS) return;
  int e    = idx & 15;
  int lane = (idx >> 4) & 31;
  int kt   = (idx >> 9) & 7;
  int blk  = idx >> 12;            // q*7 + t within slab, slabs consecutive
  int t    = blk % 7;
  int q    = (blk / 7) & 3;
  int wg   = blk / 28;
  int K    = kt * 32 + ((e < 8) ? (8 * (lane >> 4) + e) : (16 + 8 * (lane >> 4) + (e - 8)));
  int col  = wg * 64 + q * 16 + (lane & 15);   // hidden unit index
  float val;
  if (t < 3)       val = Wih[(t * 256 + col) * 256 + K];
  else if (t < 6)  val = Whh[((t - 3) * 256 + col) * 256 + K];
  else             val = Wout[col * 256 + K];
  packW[idx] = f2bf(val);
}

__global__ void k_init(const float* __restrict__ lat, unsigned short* __restrict__ xg,
                       unsigned short* __restrict__ hg, int* __restrict__ cnt) {
  int i = blockIdx.x * 256 + threadIdx.x;
  if (i < NB * HD) { xg[i] = f2bf(lat[i]); hg[i] = 0; }
  if (i == 0) __hip_atomic_store(cnt, 0, __ATOMIC_RELEASE, __HIP_MEMORY_SCOPE_AGENT);
}

// ---------------------------------------------------------------------------
// GRU persistent kernel helpers
// ---------------------------------------------------------------------------
__device__ __forceinline__ unsigned ldsoff(const void* p) {
  return (unsigned)(uintptr_t)p;     // low 32 bits of flat LDS address = LDS byte offset
}

// async DMA: global -> LDS, 16B per lane per issue (ASYNCcnt-tracked)
__device__ __forceinline__ void async_g2l(const void* gsrc, unsigned ldsBase, int bytes) {
  unsigned long long base = (unsigned long long)(uintptr_t)gsrc;
  for (int off = (int)threadIdx.x * 16; off < bytes; off += GRU_THREADS * 16) {
    unsigned l = ldsBase + (unsigned)off;
    unsigned g = (unsigned)off;
    asm volatile("global_load_async_to_lds_b128 %0, %1, %2"
                 :: "v"(l), "v"(g), "s"(base) : "memory");
  }
}
__device__ __forceinline__ void wait_async_and_sync() {
  asm volatile("s_wait_asynccnt 0" ::: "memory");
  __syncthreads();
}

// A-fragment (16x32 bf16, M=batch, K=feature) from row-major [16][256] bf16 in LDS,
// per CDNA5 ISA 16-bit A layout. Per lane-half the 8 packed dwords form two
// contiguous 16B groups -> two ds_load_b128.
__device__ __forceinline__ v16bf load_afrag(const unsigned short* act, int lane, int kt) {
  BFrag f;
  const uint4* p = (const uint4*)act;   // 16 B units; one row = 32 uint4
  int m = lane & 15, lh = lane >> 4;
  int base = m * 32 + kt * 4 + lh;      // (row*128 + kt*16 + 4*lh) dwords / 4
  f.q[0] = p[base];                     // K = {0..7}  (+8 for upper lane half)
  f.q[1] = p[base + 2];                 // K = {16..23} (+8 for upper lane half)
  return f.v;
}

// B-fragment: prepacked contiguous 32 B per lane -> two ds_load_b128
__device__ __forceinline__ v16bf load_bfrag(const unsigned short* tile, int lane, int kt) {
  BFrag f;
  const uint4* p = (const uint4*)tile + (kt * 32 + lane) * 2;
  f.q[0] = p[0];
  f.q[1] = p[1];
  return f.v;
}

__device__ __forceinline__ void grid_barrier(int* cnt, int goal) {
  __threadfence();
  __syncthreads();
  if (threadIdx.x == 0) {
    __hip_atomic_fetch_add(cnt, 1, __ATOMIC_ACQ_REL, __HIP_MEMORY_SCOPE_AGENT);
    while (__hip_atomic_load(cnt, __ATOMIC_ACQUIRE, __HIP_MEMORY_SCOPE_AGENT) < goal)
      __builtin_amdgcn_s_sleep(1);
  }
  __builtin_amdgcn_s_cluster_barrier();   // fast path if cluster-dispatched; NOP otherwise
  __syncthreads();
}

// ---------------------------------------------------------------------------
// Persistent GRU decoder: NWG=4 workgroups x 4 waves; each wave owns 16 hidden
// units end-to-end. Weights live in LDS (bf16). 1028 serial steps, two
// device-wide barriers per step. All GEMMs are v_wmma_f32_16x16x32_bf16.
// ---------------------------------------------------------------------------
__global__ void __launch_bounds__(GRU_THREADS, 1)
k_gru(const unsigned short* __restrict__ packW, unsigned short* __restrict__ xg,
      unsigned short* __restrict__ hg, int* __restrict__ cnt,
      const float* __restrict__ bih, const float* __restrict__ bhh,
      const float* __restrict__ outb, float* __restrict__ out) {
  extern __shared__ char smem[];
  unsigned short* lw = (unsigned short*)smem;                        // 229376 B weights
  unsigned short* xs = (unsigned short*)(smem + WSLAB_BYTES);        // 8 KB x  (bf16)
  unsigned short* hs = (unsigned short*)(smem + WSLAB_BYTES + ACT_BYTES); // 8 KB hid

  const int wg   = blockIdx.x;
  const int lane = threadIdx.x & 31;
  const int q    = threadIdx.x >> 5;          // wave id = hidden subgroup
  const int hb   = wg * 64;

  // stage this WG's weight slab + initial hidden state into LDS via async DMA
  async_g2l(packW + (size_t)wg * WSLAB_ELEMS, ldsoff(lw), WSLAB_BYTES);
  async_g2l(hg, ldsoff(hs), ACT_BYTES);
  wait_async_and_sync();

  const unsigned short* t_ir = lw + (q * 7 + 0) * 4096;
  const unsigned short* t_iz = lw + (q * 7 + 1) * 4096;
  const unsigned short* t_in = lw + (q * 7 + 2) * 4096;
  const unsigned short* t_hr = lw + (q * 7 + 3) * 4096;
  const unsigned short* t_hz = lw + (q * 7 + 4) * 4096;
  const unsigned short* t_hn = lw + (q * 7 + 5) * 4096;
  const unsigned short* t_o  = lw + (q * 7 + 6) * 4096;

  const int col = hb + q * 16 + (lane & 15);  // hidden unit this lane produces
  const float br   = bih[col]       + bhh[col];
  const float bz   = bih[256 + col] + bhh[256 + col];
  const float bn_i = bih[512 + col];
  const float bn_h = bhh[512 + col];
  const float bo   = outb[col];

  int epoch = 0;
  for (int step = 0; step < NCH; ++step) {
    // ---------------- phase A: gates + hidden update ----------------
    async_g2l(xg, ldsoff(xs), ACT_BYTES);     // x feedback (latent on step 0)
    wait_async_and_sync();

    v8f air = {0.f,0.f,0.f,0.f,0.f,0.f,0.f,0.f};
    v8f aiz = {0.f,0.f,0.f,0.f,0.f,0.f,0.f,0.f};
    v8f ain = {0.f,0.f,0.f,0.f,0.f,0.f,0.f,0.f};
    v8f ahr = {0.f,0.f,0.f,0.f,0.f,0.f,0.f,0.f};
    v8f ahz = {0.f,0.f,0.f,0.f,0.f,0.f,0.f,0.f};
    v8f ahn = {0.f,0.f,0.f,0.f,0.f,0.f,0.f,0.f};
#pragma unroll
    for (int kt = 0; kt < 8; ++kt) {
      v16bf ax = load_afrag(xs, lane, kt);
      v16bf ah = load_afrag(hs, lane, kt);
      air = WMMA_BF16(ax, load_bfrag(t_ir, lane, kt), air);
      aiz = WMMA_BF16(ax, load_bfrag(t_iz, lane, kt), aiz);
      ain = WMMA_BF16(ax, load_bfrag(t_in, lane, kt), ain);
      ahr = WMMA_BF16(ah, load_bfrag(t_hr, lane, kt), ahr);
      ahz = WMMA_BF16(ah, load_bfrag(t_hz, lane, kt), ahz);
      ahn = WMMA_BF16(ah, load_bfrag(t_hn, lane, kt), ahn);
    }
#pragma unroll
    for (int v = 0; v < 8; ++v) {
      int batch = v + ((lane >> 4) << 3);     // D layout: vgpr->M(+8 per lane-half)
      float r  = sigm(air[v] + ahr[v] + br);
      float z  = sigm(aiz[v] + ahz[v] + bz);
      float nn = tanhf(ain[v] + bn_i + r * (ahn[v] + bn_h));
      float hp = bf2f(hs[batch * 256 + col]);
      float hv = (1.0f - z) * nn + z * hp;
      hg[batch * 256 + col] = f2bf(hv);       // publish hidden slice
    }
    grid_barrier(cnt, ++epoch * NWG);

    // ---------------- phase B: chunk = hid_new @ out_W^T + b ----------------
    async_g2l(hg, ldsoff(hs), ACT_BYTES);     // gather full new hidden state
    wait_async_and_sync();

    v8f co = {0.f,0.f,0.f,0.f,0.f,0.f,0.f,0.f};
#pragma unroll
    for (int kt = 0; kt < 8; ++kt)
      co = WMMA_BF16(load_afrag(hs, lane, kt), load_bfrag(t_o, lane, kt), co);
#pragma unroll
    for (int v = 0; v < 8; ++v) {
      int batch = v + ((lane >> 4) << 3);
      float val = co[v] + bo;
      out[(size_t)batch * CHTOT + step * 256 + col] = val;  // fp32 output
      xg[batch * 256 + col] = f2bf(val);                    // feedback for next step
    }
    grid_barrier(cnt, ++epoch * NWG);
  }
}

// ---------------------------------------------------------------------------
// Launcher
// ---------------------------------------------------------------------------
extern "C" void kernel_launch(void* const* d_in, const int* in_sizes, int n_in,
                              void* d_out, int out_size, void* d_ws, size_t ws_size,
                              hipStream_t stream) {
  (void)in_sizes; (void)n_in; (void)out_size; (void)ws_size;
  const float* genome     = (const float*)d_in[0];
  const float* embed_W    = (const float*)d_in[1];
  const float* embed_b    = (const float*)d_in[2];
  const float* qkv_W      = (const float*)d_in[3];
  const float* qkv_b      = (const float*)d_in[4];
  const float* attn_out_W = (const float*)d_in[5];
  const float* attn_out_b = (const float*)d_in[6];
  const float* ln1_g      = (const float*)d_in[7];
  const float* ln1_b      = (const float*)d_in[8];
  const float* ln2_g      = (const float*)d_in[9];
  const float* ln2_b      = (const float*)d_in[10];
  const float* ffn_W1     = (const float*)d_in[11];
  const float* ffn_b1     = (const float*)d_in[12];
  const float* ffn_W2     = (const float*)d_in[13];
  const float* ffn_b2     = (const float*)d_in[14];
  const float* gru_Wih    = (const float*)d_in[15];
  const float* gru_Whh    = (const float*)d_in[16];
  const float* gru_bih    = (const float*)d_in[17];
  const float* gru_bhh    = (const float*)d_in[18];
  const float* out_W      = (const float*)d_in[19];
  const float* out_b      = (const float*)d_in[20];

  char* ws = (char*)d_ws;
  float* h    = (float*)(ws + 0);              // 256x256
  float* bufA = (float*)(ws + 262144);         // up to 256x1024
  float* bufB = (float*)(ws + 1310720);        // 256x256
  float* bufC = (float*)(ws + 1572864);        // 256x256
  float* lat  = (float*)(ws + 1835008);        // 16x256
  unsigned short* xg    = (unsigned short*)(ws + 1851392);
  unsigned short* hg    = (unsigned short*)(ws + 1859584);
  unsigned short* packW = (unsigned short*)(ws + 1867776);
  int* cnt = (int*)(ws + 2785280);

  dim3 t16(16, 16);
  // embedding: [256,16] @ [256,16]^T
  k_gemm<<<dim3(16, 16), t16, 0, stream>>>(genome, embed_W, embed_b, h, 256, 256, 16, 0);
  for (int l = 0; l < 2; ++l) {
    k_gemm<<<dim3(48, 16), t16, 0, stream>>>(h, qkv_W + l * 768 * 256, qkv_b + l * 768,
                                             bufA, 256, 768, 256, 0);
    k_attention<<<64, t16, 0, stream>>>(bufA, bufB);
    k_gemm<<<dim3(16, 16), t16, 0, stream>>>(bufB, attn_out_W + l * 65536, attn_out_b + l * 256,
                                             bufC, 256, 256, 256, 0);
    k_add_ln<<<256, 256, 0, stream>>>(h, bufC, ln1_g + l * 256, ln1_b + l * 256);
    k_gemm<<<dim3(64, 16), t16, 0, stream>>>(h, ffn_W1 + l * 262144, ffn_b1 + l * 1024,
                                             bufA, 256, 1024, 256, 1);
    k_gemm<<<dim3(16, 16), t16, 0, stream>>>(bufA, ffn_W2 + l * 262144, ffn_b2 + l * 256,
                                             bufC, 256, 256, 1024, 0);
    k_add_ln<<<256, 256, 0, stream>>>(h, bufC, ln2_g + l * 256, ln2_b + l * 256);
  }
  k_mean<<<16, 256, 0, stream>>>(h, lat);
  k_pack<<<1792, 256, 0, stream>>>(gru_Wih, gru_Whh, out_W, packW);
  k_init<<<16, 256, 0, stream>>>(lat, xg, hg, cnt);
  k_gru<<<NWG, GRU_THREADS, SMEM_BYTES, stream>>>(packW, xg, hg, cnt,
                                                  gru_bih, gru_bhh, out_b, (float*)d_out);
}
